// TitansBlock_77326591197225
// MI455X (gfx1250) — compile-verified, matrix operand order checked by reading, not run on hardware
//
#include <hip/hip_runtime.h>
#include <cmath>

#define AS3 __attribute__((address_space(3)))

// ---------------------------------------------------------------------------
// Types / helpers
// ---------------------------------------------------------------------------
typedef __attribute__((ext_vector_type(16))) __bf16 v16bf;
typedef __attribute__((ext_vector_type(8)))  float  v8f;

__device__ __forceinline__ unsigned short f2bf(float f) {
  unsigned int u = __builtin_bit_cast(unsigned int, f);
  unsigned int r = u + 0x7FFFu + ((u >> 16) & 1u);   // round-to-nearest-even
  return (unsigned short)(r >> 16);
}

__device__ __forceinline__ unsigned lds_off(const void* ptr) {
  return (unsigned)(unsigned long long)(AS3 const void*)ptr;
}

// ---------------------------------------------------------------------------
// Batched BF16 WMMA GEMM, double-buffered async-DMA pipeline:
//   A: [M][K] row-major (lda), B: [N][K] row-major (ldb)  (i.e. column-major B)
//   pre  = alpha*acc + bias[n] + beta*addC[m,n]
//   C    = pre                  (f32, optional)
//   Cbf  = act(pre)             (bf16, optional; act: 0=none 1=gelu 2=silu)
// Batch z -> (zo=z/zdiv, zi=z%zdiv) with independent outer/inner strides.
// CONTRACT (all call sites satisfy): K%32==0, M%16==0, N%16==0,
//   base pointers 16B aligned, lda/ldb multiples of 8 elements.
// ---------------------------------------------------------------------------
struct GemmP {
  const unsigned short* A;
  const unsigned short* B;
  float* C;
  unsigned short* Cbf;
  const float* bias;
  const float* addC;
  long long sAo, sAi, sBo, sBi, sCo, sCi, sCbo, sCbi, sDo, sDi;
  int zdiv;
  int M, N, K, lda, ldb, ldc, ldcbf, ldadd;
  int act;
  float alpha, beta;
};

#define BM 64
#define BN 128
#define BK 32
#define LDS_K 40   // row stride (elements); 80B keeps 16B alignment

__global__ __launch_bounds__(256) void gemm_wmma_bf16(GemmP p) {
  __shared__ unsigned short sA[2][BM * LDS_K];  // 2 x  5 KB
  __shared__ unsigned short sB[2][BN * LDS_K];  // 2 x 10 KB

  const int zo = (int)blockIdx.z / p.zdiv;
  const int zi = (int)blockIdx.z % p.zdiv;
  const unsigned short* A = p.A + (long long)zo * p.sAo + (long long)zi * p.sAi;
  const unsigned short* B = p.B + (long long)zo * p.sBo + (long long)zi * p.sBi;

  const int tid  = threadIdx.x;
  const int lane = tid & 31;
  const int wave = tid >> 5;
  const int wm   = wave >> 1;           // 0..3 -> row subtile
  const int wn   = wave & 1;            // 0..1 -> col half
  const int m0   = (int)blockIdx.y * BM;
  const int n0   = (int)blockIdx.x * BN;
  const int lm   = lane & 15;
  const int koff = (lane < 16) ? 0 : 8; // CDNA5 16-bit operand lane split

  // cooperative-DMA coordinates (K-contiguous for both tiles)
  const int ar = tid >> 2, akk = (tid & 3) * 8;   // A: 16B/thread
  const int bn = tid >> 1, bkk = (tid & 1) * 16;  // B: 32B/thread
  const int gmA = m0 + ar;
  const int gnB = n0 + bn;
  const bool okA = gmA < p.M;
  const bool okB = gnB < p.N;

  unsigned short* dA0 = &sA[0][ar * LDS_K + akk];
  unsigned short* dA1 = &sA[1][ar * LDS_K + akk];
  unsigned short* dB0 = &sB[0][bn * LDS_K + bkk];
  unsigned short* dB1 = &sB[1][bn * LDS_K + bkk];
  const unsigned ldsA[2] = { lds_off(dA0), lds_off(dA1) };
  const unsigned ldsB[2] = { lds_off(dB0), lds_off(dB1) };
  const unsigned short* gAp = &A[(long long)gmA * p.lda + akk];
  const unsigned short* gBp = &B[(long long)gnB * p.ldb + bkk];

  // one-time zero-fill of OOB rows in both buffers (never overwritten by DMA)
  if (!okA) { uint4 z = {0u,0u,0u,0u}; *(uint4*)dA0 = z; *(uint4*)dA1 = z; }
  if (!okB) { uint4 z = {0u,0u,0u,0u};
              ((uint4*)dB0)[0] = z; ((uint4*)dB0)[1] = z;
              ((uint4*)dB1)[0] = z; ((uint4*)dB1)[1] = z; }

  // prologue: issue tile 0 into buffer 0
  if (okA)
    asm volatile("global_load_async_to_lds_b128 %0, %1, off"
                 :: "v"(ldsA[0]), "v"((unsigned long long)gAp) : "memory");
  if (okB)
    asm volatile("global_load_async_to_lds_b128 %0, %1, off\n\t"
                 "global_load_async_to_lds_b128 %0, %1, off offset:16"
                 :: "v"(ldsB[0]), "v"((unsigned long long)gBp) : "memory");

  v8f acc[4] = {};
  const int nk = p.K / BK;

  for (int t = 0; t < nk; ++t) {
    const int buf = t & 1;
    // wait for this wave's DMA into `buf`, then make it visible block-wide.
    // The same barrier also retires all reads of buf^1 from iteration t-1,
    // making it safe to overwrite below.
    asm volatile("s_wait_asynccnt 0x0" ::: "memory");
    __syncthreads();

    // issue next tile into the other buffer; overlaps with WMMA below
    if (t + 1 < nk) {
      const long long k1 = (long long)(t + 1) * BK;
      if (okA)
        asm volatile("global_load_async_to_lds_b128 %0, %1, off"
                     :: "v"(ldsA[buf ^ 1]), "v"((unsigned long long)(gAp + k1)) : "memory");
      if (okB)
        asm volatile("global_load_async_to_lds_b128 %0, %1, off\n\t"
                     "global_load_async_to_lds_b128 %0, %1, off offset:16"
                     :: "v"(ldsB[buf ^ 1]), "v"((unsigned long long)(gBp + k1)) : "memory");
    }

    // ---- A fragment: K-pairs contiguous -> ds_load_b128 pairs ----
    union { unsigned int u[8]; v16bf v; } af;
    const unsigned short* arow = &sA[buf][(wm * 16 + lm) * LDS_K];
    #pragma unroll
    for (int i = 0; i < 4; ++i) {
      af.u[i]     = *(const unsigned int*)&arow[koff + 2 * i];
      af.u[4 + i] = *(const unsigned int*)&arow[koff + 16 + 2 * i];
    }

    // ---- 4 col-subtiles per wave ----
    #pragma unroll
    for (int s = 0; s < 4; ++s) {
      union { unsigned int u[8]; v16bf v; } bfr;
      const unsigned short* brow = &sB[buf][(wn * 64 + s * 16 + lm) * LDS_K];
      #pragma unroll
      for (int i = 0; i < 4; ++i) {
        bfr.u[i]     = *(const unsigned int*)&brow[koff + 2 * i];
        bfr.u[4 + i] = *(const unsigned int*)&brow[koff + 16 + 2 * i];
      }
      acc[s] = __builtin_amdgcn_wmma_f32_16x16x32_bf16(
          false, af.v, false, bfr.v, (short)0, acc[s], false, false);
    }
  }

  // ---- epilogue (compact: unified sigmoid-based activation) ----
  float* Cz = p.C ? p.C + (long long)zo * p.sCo + (long long)zi * p.sCi : nullptr;
  unsigned short* Cbz = p.Cbf ? p.Cbf + (long long)zo * p.sCbo + (long long)zi * p.sCbi : nullptr;
  const float* Az = p.addC ? p.addC + (long long)zo * p.sDo + (long long)zi * p.sDi : nullptr;

  const int colBase = n0 + wn * 64 + lm;
  bool okc[4];
  #pragma unroll
  for (int s = 0; s < 4; ++s) okc[s] = (colBase + s * 16) < p.N;

  #pragma unroll
  for (int r = 0; r < 8; ++r) {
    const int gm = m0 + wm * 16 + r + ((lane < 16) ? 0 : 8);
    if (gm < p.M) {
      const long long rowC  = (long long)gm * p.ldc;
      const long long rowCb = (long long)gm * p.ldcbf;
      const long long rowD  = (long long)gm * p.ldadd;
      #pragma unroll
      for (int s = 0; s < 4; ++s) {
        if (okc[s]) {
          const int gn = colBase + s * 16;
          float v = p.alpha * acc[s][r];
          if (p.bias) v += p.bias[gn];
          if (Az)     v += p.beta * Az[rowD + gn];
          if (Cz)     Cz[rowC + gn] = v;
          if (Cbz) {
            float o = v;
            if (p.act) {
              // gelu(tanh) == v * sigmoid(1.59577*(v + 0.044715*v^3)); silu == v*sigmoid(v)
              float y = (p.act == 1) ? 1.5957691f * (v + 0.044715f * v * v * v) : v;
              o = v / (1.f + __expf(-y));
            }
            Cbz[rowCb + gn] = f2bf(o);
          }
        }
      }
    }
  }
}

// ---------------------------------------------------------------------------
// LayerNorm -> bf16 (C=384 fixed), flexible output placement
// ---------------------------------------------------------------------------
__global__ __launch_bounds__(128) void ln_kernel(const float* x, const float* gw,
                                                 const float* bw, unsigned short* out,
                                                 int T, long long outBatchStride,
                                                 int outRowOff) {
  __shared__ float red[128];
  int row = blockIdx.x;
  const float* xr = x + (long long)row * 384;
  float s = 0.f;
  for (int c = threadIdx.x; c < 384; c += 128) s += xr[c];
  red[threadIdx.x] = s; __syncthreads();
  for (int o = 64; o > 0; o >>= 1) { if (threadIdx.x < o) red[threadIdx.x] += red[threadIdx.x + o]; __syncthreads(); }
  float mu = red[0] / 384.f; __syncthreads();
  float v = 0.f;
  for (int c = threadIdx.x; c < 384; c += 128) { float d = xr[c] - mu; v += d * d; }
  red[threadIdx.x] = v; __syncthreads();
  for (int o = 64; o > 0; o >>= 1) { if (threadIdx.x < o) red[threadIdx.x] += red[threadIdx.x + o]; __syncthreads(); }
  float rstd = rsqrtf(red[0] / 384.f + 1e-5f);
  unsigned short* o = out + (long long)(row / T) * outBatchStride
                          + (long long)(outRowOff + row % T) * 384;
  for (int c = threadIdx.x; c < 384; c += 128)
    o[c] = f2bf((xr[c] - mu) * rstd * gw[c] + bw[c]);
}

// ---------------------------------------------------------------------------
// Prefix-LM masked softmax. Mask is a contiguous range:
//   allowed keys for query q: [0, PL) if q < PL, else [0, q].
// ---------------------------------------------------------------------------
__global__ void attn_softmax(const float* S, unsigned short* P,
                             int tReal, int PL, int Tpad) {
  int z = blockIdx.x, q = threadIdx.x;
  const float* Sr = S + ((long long)z * Tpad + q) * Tpad;
  unsigned short* Pr = P + ((long long)z * Tpad + q) * Tpad;
  int kmax = (q >= tReal) ? 0 : ((q < PL) ? PL : q + 1);
  float mx = -1e30f;
  for (int k = 0; k < kmax; ++k) mx = fmaxf(mx, Sr[k]);
  float sum = 0.f;
  for (int k = 0; k < kmax; ++k) sum += __expf(Sr[k] - mx);
  float inv = (kmax > 0) ? 1.0f / sum : 0.f;
  for (int k = 0; k < kmax; ++k) Pr[k] = f2bf(__expf(Sr[k] - mx) * inv);
  for (int k = kmax; k < Tpad; ++k) Pr[k] = 0;
}

// ---------------------------------------------------------------------------
// coeff[i] = -LR(1-MOM) * sum_{j=i}^{127} 0.999^{127-j} * 0.9^{j-i}
// ---------------------------------------------------------------------------
__global__ void coeff_kernel(float* coeff) {
  int i = threadIdx.x;
  if (i < 128) {
    float a = 1.f;
    for (int k = 0; k < 127 - i; ++k) a *= 0.999f;   // 0.999^(127-i)
    float s = 0.f;
    for (int j = i; j <= 127; ++j) { s += a; a = a / 0.999f * 0.9f; }
    coeff[i] = -0.001f * s;                           // scale = -LR*(1-MOM)
  }
}

// err_s stored [b][c][t]: out[i] = coeff[t]*err[(b*128+t)*384+c]
__global__ void errs_kernel(const float* err, const float* coeff,
                            unsigned short* out, int n) {
  int i = blockIdx.x * 256 + threadIdx.x;
  if (i < n) {
    int b = i / (384 * 128), c = (i / 128) % 384, t = i % 128;
    out[i] = f2bf(err[((long long)(b * 128 + t)) * 384 + c] * coeff[t]);
  }
}

// dh_s stored [b][h][t]: out[i] = coeff[t] * dhpre[t,h] * silu'(hh[t,h])
__global__ void dhs_kernel(const float* dhpre, const float* hh, const float* coeff,
                           unsigned short* out, int n) {
  int i = blockIdx.x * 256 + threadIdx.x;
  if (i < n) {
    int b = i / (768 * 128), h = (i / 128) % 768, t = i % 128;
    long long src = ((long long)(b * 128 + t)) * 768 + h;
    float hv = hh[src];
    float sg = 1.f / (1.f + __expf(-hv));
    out[i] = f2bf(dhpre[src] * (sg * (1.f + hv * (1.f - sg))) * coeff[t]);
  }
}

// aT stored [b][h][t]: silu(hh[t,h])
__global__ void at_kernel(const float* hh, unsigned short* out, int n) {
  int i = blockIdx.x * 256 + threadIdx.x;
  if (i < n) {
    int b = i / (768 * 128), h = (i / 128) % 768, t = i % 128;
    float v = hh[((long long)(b * 128 + t)) * 768 + h];
    out[i] = f2bf(v / (1.f + __expf(-v)));
  }
}

// vT stored [z][d][key] (z = b*6+h) from qkv [b][key][1152] V block
__global__ void vt_kernel(const unsigned short* qkv, unsigned short* vt, int n) {
  int i = blockIdx.x * 256 + threadIdx.x;
  if (i < n) {
    int z = i / (64 * 160), d = (i / 160) % 64, key = i % 160;
    int b = z / 6, h = z % 6;
    vt[i] = qkv[((long long)(b * 160 + key)) * 1152 + 768 + h * 64 + d];
  }
}

// f32 [K,N] -> bf16 transposed [N,K] (+ optional f32 transposed copy)
__global__ void cvt_t(const float* in, unsigned short* outb, float* outf,
                      int K, int N) {
  int i = blockIdx.x * 256 + threadIdx.x;
  if (i < N * K) {
    int n = i / K, k = i % K;
    float v = in[(long long)k * N + n];
    outb[i] = f2bf(v);
    if (outf) outf[i] = v;
  }
}

// f32 -> bf16 plain copy
__global__ void cvt_bf16(const float* in, unsigned short* out, int n) {
  int i = blockIdx.x * 256 + threadIdx.x;
  if (i < n) out[i] = f2bf(in[i]);
}

// persist rows -> bf16 into seq rows 0..3 per batch; zero pad rows 148..159
__global__ void prep_seq(const float* persist, unsigned short* seq) {
  int tid = blockIdx.x * 256 + threadIdx.x;
  for (int b = 0; b < 2; ++b) {
    unsigned short* s = seq + b * 160 * 384;
    if (tid < 4 * 384)  s[tid] = f2bf(persist[tid]);
    if (tid < 12 * 384) s[148 * 384 + tid] = 0;
  }
}

// ---------------------------------------------------------------------------
// Host side
// ---------------------------------------------------------------------------
static GemmP mk(const unsigned short* A, const unsigned short* B,
                int M, int N, int K, int lda, int ldb) {
  GemmP p{};
  p.A = A; p.B = B; p.M = M; p.N = N; p.K = K;
  p.lda = lda; p.ldb = ldb; p.ldc = N; p.ldcbf = N; p.ldadd = N;
  p.zdiv = 1; p.alpha = 1.f; p.beta = 1.f;
  return p;
}
static void gemmL(hipStream_t st, const GemmP& p, int batches) {
  dim3 g((unsigned)((p.N + BN - 1) / BN), (unsigned)((p.M + BM - 1) / BM), (unsigned)batches);
  gemm_wmma_bf16<<<g, dim3(256), 0, st>>>(p);
}

extern "C" void kernel_launch(void* const* d_in, const int* in_sizes, int n_in,
                              void* d_out, int out_size, void* d_ws, size_t ws_size,
                              hipStream_t stream) {
  (void)in_sizes; (void)n_in; (void)out_size;
  const float* x        = (const float*)d_in[0];
  const float* c_attn_w = (const float*)d_in[1];
  const float* c_attn_b = (const float*)d_in[2];
  const float* c_proj_w = (const float*)d_in[3];
  const float* c_proj_b = (const float*)d_in[4];
  const float* ln1_g    = (const float*)d_in[5];
  const float* ln1_b    = (const float*)d_in[6];
  const float* ln2_g    = (const float*)d_in[7];
  const float* ln2_b    = (const float*)d_in[8];
  const float* fc_w     = (const float*)d_in[9];
  const float* fc_b     = (const float*)d_in[10];
  const float* pr_w     = (const float*)d_in[11];
  const float* pr_b     = (const float*)d_in[12];
  const float* mem_w1   = (const float*)d_in[13];
  const float* mem_w2   = (const float*)d_in[14];
  const float* wk       = (const float*)d_in[15];
  const float* wv       = (const float*)d_in[16];
  const float* mq       = (const float*)d_in[17];
  const float* persist  = (const float*)d_in[18];
  float* out = (float*)d_out;

  // ---- workspace bump allocator (256B aligned blocks) ----
  char* base = (char*)d_ws;
  size_t off = 0;
  auto takeF = [&](size_t n) { float* r = (float*)(base + off);
                               off = (off + n * 4 + 255) & ~(size_t)255; return r; };
  auto takeH = [&](size_t n) { unsigned short* r = (unsigned short*)(base + off);
                               off = (off + n * 2 + 255) & ~(size_t)255; return r; };

  float* coeff = takeF(128);
  // transposed [N][K] bf16 weight copies (B operands)
  unsigned short* cat_bf = takeH(1152 * 384);
  unsigned short* cpt_bf = takeH(384 * 384);
  unsigned short* fct_bf = takeH(1536 * 384);
  unsigned short* prt_bf = takeH(384 * 1536);
  unsigned short* w1t_bf = takeH(768 * 384);
  unsigned short* w2t_bf = takeH(384 * 768);
  unsigned short* wkt_bf = takeH(384 * 384);
  unsigned short* wvt_bf = takeH(384 * 384);
  unsigned short* w2_bf  = takeH(768 * 384);     // plain copy (dh GEMM B)
  unsigned short* mq_bf  = takeH(16 * 384);
  float* w1t_f = takeF(768 * 384);               // f32 transposed (addC)
  float* w2t_f = takeF(384 * 768);
  unsigned short* seq_bf = takeH(2 * 160 * 384);
  unsigned short* hid_bf = takeH(16 * 768);
  unsigned short* qkv_bf = takeH(2 * 160 * 1152);
  float* S = takeF(12 * 160 * 160);
  unsigned short* P_bf = takeH(12 * 160 * 160);
  unsigned short* vt_bf = takeH(12 * 64 * 160);
  unsigned short* y_bf = takeH(2 * 160 * 384);
  float* h1 = takeF(256 * 384);
  unsigned short* h2n_bf = takeH(256 * 384);
  unsigned short* g_bf = takeH(256 * 1536);
  float* h2 = takeF(256 * 384);
  unsigned short* h2_bf = takeH(256 * 384);
  unsigned short* km_bf = takeH(256 * 384);
  unsigned short* kmT_bf = takeH(2 * 384 * 128);
  float* vm = takeF(256 * 384);
  float* hh = takeF(256 * 768);
  unsigned short* a_bf = takeH(256 * 768);
  unsigned short* aT_bf = takeH(2 * 768 * 128);
  float* err = takeF(256 * 384);
  unsigned short* err_bf = takeH(256 * 384);
  unsigned short* errs_bf = takeH(2 * 384 * 128);
  float* dhpre = takeF(256 * 768);
  unsigned short* dhs_bf = takeH(2 * 768 * 128);
  unsigned short* w1fT_bf = takeH(2 * 768 * 384);
  unsigned short* w2fT_bf = takeH(2 * 384 * 768);
  unsigned short* mh_bf = takeH(2 * 128 * 768);
  if (off > ws_size) return;

  const float decayT = (float)std::pow(0.999, 128.0);  // 0.999^T

  // ---- constants + (transposed) bf16 weights ----
  coeff_kernel<<<1, 128, 0, stream>>>(coeff);
  auto cvtT = [&](const float* in, unsigned short* ob, float* of, int K, int N) {
    cvt_t<<<(N * K + 255) / 256, 256, 0, stream>>>(in, ob, of, K, N);
  };
  cvtT(c_attn_w, cat_bf, nullptr, 384, 1152);
  cvtT(c_proj_w, cpt_bf, nullptr, 384, 384);
  cvtT(fc_w, fct_bf, nullptr, 384, 1536);
  cvtT(pr_w, prt_bf, nullptr, 1536, 384);
  cvtT(mem_w1, w1t_bf, w1t_f, 384, 768);
  cvtT(mem_w2, w2t_bf, w2t_f, 768, 384);
  cvtT(wk, wkt_bf, nullptr, 384, 384);
  cvtT(wv, wvt_bf, nullptr, 384, 384);
  cvt_bf16<<<(768 * 384 + 255) / 256, 256, 0, stream>>>(mem_w2, w2_bf, 768 * 384);
  cvt_bf16<<<(16 * 384 + 255) / 256, 256, 0, stream>>>(mq, mq_bf, 16 * 384);
  prep_seq<<<18, 256, 0, stream>>>(persist, seq_bf);

  // ---- ln1 -> xn (rows 20..147 of each batch's seq) ----
  ln_kernel<<<256, 128, 0, stream>>>(x, ln1_g, ln1_b, seq_bf, 128, 160 * 384LL, 20);

  // ---- retrieval: retr = silu(mq@w1)@w2 -> seq rows 4..19 (both batches) ----
  { GemmP p = mk(mq_bf, w1t_bf, 16, 768, 384, 384, 384);
    p.Cbf = hid_bf; p.act = 2; gemmL(stream, p, 1); }
  { GemmP p = mk(hid_bf, w2t_bf, 16, 384, 768, 768, 768);
    p.Cbf = seq_bf + 4 * 384; p.sCbo = 160 * 384; gemmL(stream, p, 2); }

  // ---- qkv = seq @ c_attn_w + b ----
  { GemmP p = mk(seq_bf, cat_bf, 320, 1152, 384, 384, 384);
    p.bias = c_attn_b; p.Cbf = qkv_bf; gemmL(stream, p, 1); }

  // ---- S = Q K^T / 8, batched over (b,h); B = K[key][dim] is natural ----
  { GemmP p = mk(qkv_bf, qkv_bf + 384, 160, 160, 64, 1152, 1152);
    p.alpha = 0.125f; p.C = S;
    p.zdiv = 6; p.sAo = 160 * 1152; p.sAi = 64; p.sBo = 160 * 1152; p.sBi = 64;
    p.sCo = 6 * 160 * 160; p.sCi = 160 * 160; gemmL(stream, p, 12); }

  attn_softmax<<<12, 160, 0, stream>>>(S, P_bf, 148, 20, 160);
  vt_kernel<<<(12 * 64 * 160 + 255) / 256, 256, 0, stream>>>(qkv_bf, vt_bf, 12 * 64 * 160);

  // ---- y = P @ V (B = V^T [dim][key]) -> interleaved [b,160,384] at col h*64 ----
  { GemmP p = mk(P_bf, vt_bf, 160, 64, 160, 160, 160);
    p.Cbf = y_bf; p.ldcbf = 384;
    p.zdiv = 6; p.sAo = 6 * 160 * 160; p.sAi = 160 * 160;
    p.sBo = 6 * 64 * 160; p.sBi = 64 * 160;
    p.sCbo = 160 * 384; p.sCbi = 64; gemmL(stream, p, 12); }

  // ---- h1 = x + proj(y[suffix]) ----
  { GemmP p = mk(y_bf + 20 * 384, cpt_bf, 128, 384, 384, 384, 384);
    p.bias = c_proj_b; p.addC = x; p.C = h1;
    p.sAo = 160 * 384; p.sDo = 128 * 384; p.sCo = 128 * 384; gemmL(stream, p, 2); }

  // ---- MLP: h2 = h1 + gelu(ln2(h1)@fc)@pr ----
  ln_kernel<<<256, 128, 0, stream>>>(h1, ln2_g, ln2_b, h2n_bf, 128, 128 * 384LL, 0);
  { GemmP p = mk(h2n_bf, fct_bf, 256, 1536, 384, 384, 384);
    p.bias = fc_b; p.act = 1; p.Cbf = g_bf; gemmL(stream, p, 1); }
  { GemmP p = mk(g_bf, prt_bf, 256, 384, 1536, 1536, 1536);
    p.bias = pr_b; p.addC = h1; p.C = h2; p.Cbf = h2_bf; gemmL(stream, p, 1); }

  // ---- Titans memory ----
  { GemmP p = mk(seq_bf + 20 * 384, wkt_bf, 128, 384, 384, 384, 384);
    p.Cbf = km_bf; p.sAo = 160 * 384; p.sCbo = 128 * 384; gemmL(stream, p, 2); }
  { GemmP p = mk(wkt_bf, seq_bf + 20 * 384, 384, 128, 384, 384, 384);
    p.Cbf = kmT_bf; p.ldcbf = 128; p.sBo = 160 * 384; p.sCbo = 384 * 128;
    gemmL(stream, p, 2); }
  { GemmP p = mk(seq_bf + 20 * 384, wvt_bf, 128, 384, 384, 384, 384);
    p.C = vm; p.sAo = 160 * 384; p.sCo = 128 * 384; gemmL(stream, p, 2); }
  { GemmP p = mk(km_bf, w1t_bf, 256, 768, 384, 384, 384);
    p.C = hh; p.Cbf = a_bf; p.act = 2; gemmL(stream, p, 1); }
  { GemmP p = mk(a_bf, w2t_bf, 256, 384, 768, 768, 768);
    p.addC = vm; p.beta = -1.f; p.C = err; p.Cbf = err_bf; gemmL(stream, p, 1); }
  { GemmP p = mk(err_bf, w2_bf, 256, 768, 384, 384, 384);
    p.C = dhpre; gemmL(stream, p, 1); }

  errs_kernel<<<(2 * 384 * 128 + 255) / 256, 256, 0, stream>>>(err, coeff, errs_bf, 2 * 384 * 128);
  dhs_kernel<<<(2 * 768 * 128 + 255) / 256, 256, 0, stream>>>(dhpre, hh, coeff, dhs_bf, 2 * 768 * 128);
  at_kernel<<<(2 * 768 * 128 + 255) / 256, 256, 0, stream>>>(hh, aT_bf, 2 * 768 * 128);

  // ---- scans collapsed (transposed outputs, ready as B operands) ----
  { GemmP p = mk(dhs_bf, kmT_bf, 768, 384, 128, 128, 128);
    p.addC = w1t_f; p.beta = decayT; p.Cbf = w1fT_bf;
    p.sAo = 768 * 128; p.sBo = 384 * 128; p.sCbo = 768 * 384; gemmL(stream, p, 2); }
  { GemmP p = mk(errs_bf, aT_bf, 384, 768, 128, 128, 128);
    p.addC = w2t_f; p.beta = decayT; p.Cbf = w2fT_bf;
    p.sAo = 384 * 128; p.sBo = 768 * 128; p.sCbo = 384 * 768; gemmL(stream, p, 2); }

  // ---- out = h2 + silu(h2 @ w1f) @ w2f ----
  { GemmP p = mk(h2_bf, w1fT_bf, 128, 768, 384, 384, 384);
    p.act = 2; p.Cbf = mh_bf;
    p.sAo = 128 * 384; p.sBo = 768 * 384; p.sCbo = 128 * 768; gemmL(stream, p, 2); }
  { GemmP p = mk(mh_bf, w2fT_bf, 128, 384, 768, 768, 768);
    p.addC = h2; p.C = out;
    p.sAo = 128 * 768; p.sBo = 384 * 768; p.sDo = 128 * 384; p.sCo = 128 * 384;
    gemmL(stream, p, 2); }
}
